// CausalSelfAttention_6030134084255
// MI455X (gfx1250) — matrix operand order repaired and checked by reading, not compiled, not run on hardware
//
#include <hip/hip_runtime.h>

typedef _Float16 f16;
typedef __attribute__((ext_vector_type(8)))  _Float16 v8h;
typedef __attribute__((ext_vector_type(16))) _Float16 v16h;
typedef __attribute__((ext_vector_type(8)))  float    v8f;
typedef __attribute__((ext_vector_type(4)))  int      v4i;

#define B_  2
#define T_  2048
#define C_  1024
#define H_  16
#define D_  64
#define C3_ 3072

#if __has_builtin(__builtin_amdgcn_global_load_async_to_lds_b128)
#define ASYNC_LDS 1
#else
#define ASYNC_LDS 0
#endif

typedef __attribute__((address_space(1))) v4i GV4;  // global int4
typedef __attribute__((address_space(3))) v4i LV4;  // LDS int4

__device__ __forceinline__ GV4* to_global_v4(const void* p) {
  return reinterpret_cast<GV4*>(reinterpret_cast<uintptr_t>(p));
}
__device__ __forceinline__ LV4* to_lds_v4(const void* p) {
  // flat LDS addresses keep the byte offset in the low 32 bits
  return reinterpret_cast<LV4*>((unsigned)reinterpret_cast<uintptr_t>(p));
}
__device__ __forceinline__ void wait_async0() {
#if __has_builtin(__builtin_amdgcn_s_wait_asynccnt)
  __builtin_amdgcn_s_wait_asynccnt(0);
#else
  asm volatile("s_wait_asynccnt 0x0" ::: "memory");
#endif
}

__device__ __forceinline__ v16h hcat(v8h lo, v8h hi) {
  v16h r;
#pragma unroll
  for (int i = 0; i < 8; ++i) { r[i] = lo[i]; r[i + 8] = hi[i]; }
  return r;
}

__device__ __forceinline__ v8f wmma16(v16h a, v16h b, v8f c) {
  // D = A(16x32 f16) * B(32x16 f16) + C(16x16 f32)
  return __builtin_amdgcn_wmma_f32_16x16x32_f16(false, a, false, b, (short)0, c,
                                                false, false);
}

// ---------------------------------------------------------------------------
// fp32 -> f16 elementwise convert
// ---------------------------------------------------------------------------
__global__ void cvt_f16_kernel(const float* __restrict__ in, f16* __restrict__ out,
                               int n) {
  int i = blockIdx.x * blockDim.x + threadIdx.x;
  if (i < n) out[i] = (f16)in[i];
}

// fp32 W[K,N] -> f16 Wt[N,K] (transpose so GEMM B-operand loads are contiguous)
__global__ void cvt_transpose_kernel(const float* __restrict__ W,
                                     f16* __restrict__ Wt, int K, int N) {
  int i = blockIdx.x * blockDim.x + threadIdx.x; // index into Wt
  if (i < N * K) {
    int nn = i / K;
    int k  = i - nn * K;
    Wt[i] = (f16)W[(size_t)k * N + nn];
  }
}

// ---------------------------------------------------------------------------
// GEMM: out[M,N] = A[M,K] * Bt[N,K]^T + bias[N]
// One wave per 32x64 output tile (2x4 WMMA subtiles): per 32-wide K chunk we
// load 6 operands (192 B/lane) and issue 8 WMMAs -> ~21 FLOP/byte.
// ---------------------------------------------------------------------------
__global__ void __launch_bounds__(256)
gemm_f16_kernel(const f16* __restrict__ A, const f16* __restrict__ Bt,
                const float* __restrict__ bias, f16* out_h, float* out_f,
                int M, int N, int K) {
  const int lane = threadIdx.x & 31;
  const int wave = threadIdx.x >> 5;
  const int row  = lane & 15;   // A row / B column within a 16-subtile
  const int hi   = lane >> 4;   // wave half

  const int tiles_n = N >> 6;   // 64-wide N tiles
  const int tiles_m = M >> 5;   // 32-tall M tiles
  const int tid = blockIdx.x * 8 + wave;
  if (tid >= tiles_m * tiles_n) return;
  const int mt = tid / tiles_n;
  const int nt = tid - mt * tiles_n;

  const f16* arow[2];
  arow[0] = A + (size_t)(mt * 32 + row) * K;
  arow[1] = arow[0] + (size_t)16 * K;
  const f16* brow[4];
#pragma unroll
  for (int ni = 0; ni < 4; ++ni)
    brow[ni] = Bt + (size_t)(nt * 64 + ni * 16 + row) * K;

  v8f acc[2][4];
#pragma unroll
  for (int mi = 0; mi < 2; ++mi)
#pragma unroll
    for (int ni = 0; ni < 4; ++ni) { v8f z = {}; acc[mi][ni] = z; }

  for (int k0 = 0; k0 < K; k0 += 32) {
    __builtin_prefetch(arow[0] + k0 + 512, 0, 0);
    __builtin_prefetch(brow[0] + k0 + 512, 0, 0);
    v16h a[2], b[4];
#pragma unroll
    for (int mi = 0; mi < 2; ++mi) {
      // A 16x32 layout: half 'hi' holds K = k0+hi*8+{0..7} and k0+16+hi*8+{0..7}
      v8h lo  = *(const v8h*)(arow[mi] + k0 + hi * 8);
      v8h hi8 = *(const v8h*)(arow[mi] + k0 + 16 + hi * 8);
      a[mi] = hcat(lo, hi8);
    }
#pragma unroll
    for (int ni = 0; ni < 4; ++ni) {
      // B 32x16 layout: half 'hi' holds K = k0+hi*8+{0..7} and k0+16+hi*8+{0..7}
      v8h lo  = *(const v8h*)(brow[ni] + k0 + hi * 8);
      v8h hi8 = *(const v8h*)(brow[ni] + k0 + 16 + hi * 8);
      b[ni] = hcat(lo, hi8);
    }
#pragma unroll
    for (int mi = 0; mi < 2; ++mi)
#pragma unroll
      for (int ni = 0; ni < 4; ++ni)
        acc[mi][ni] = wmma16(a[mi], b[ni], acc[mi][ni]);
  }

#pragma unroll
  for (int ni = 0; ni < 4; ++ni) {
    const float bv = bias[nt * 64 + ni * 16 + row];
#pragma unroll
    for (int mi = 0; mi < 2; ++mi)
#pragma unroll
      for (int r = 0; r < 8; ++r) {
        // C/D layout: element (M = r + 8*hi, N = row)
        const size_t idx =
            (size_t)(mt * 32 + mi * 16 + r + 8 * hi) * N + nt * 64 + ni * 16 + row;
        const float v = acc[mi][ni][r] + bv;
        if (out_h) out_h[idx] = (f16)v;
        else       out_f[idx] = v;
      }
  }
}

// ---------------------------------------------------------------------------
// Flash attention: one wave per 16-query tile of one (b,h); 4 waves / block.
// qkv: f16 [B*T, 3C] rows (q | k | v per token). att: f16 [B*T, C].
// V tiles are staged into LDS with GLOBAL_LOAD_ASYNC_TO_LDS_B128 (ASYNCcnt).
// ---------------------------------------------------------------------------
__global__ void __launch_bounds__(128)
flash_attn_kernel(const f16* __restrict__ qkv, f16* __restrict__ att) {
  __shared__ __align__(16) f16 ldsP[4][16 * 32];  // P tile (queries x 32 keys)
  __shared__ __align__(16) f16 ldsV[4][32 * 64];  // V tile (32 keys x D)

  const int lane = threadIdx.x & 31;
  const int wave = threadIdx.x >> 5;
  const int row  = lane & 15;
  const int hi   = lane >> 4;

  const int gtile = blockIdx.x * 4 + wave;      // global q-tile id
  const int qt = gtile & (T_ / 16 - 1);         // T_/16 == 128
  const int bh = gtile >> 7;
  const int h  = bh & (H_ - 1);
  const int b  = bh >> 4;
  const int q0 = qt * 16;
  const size_t baseRow = (size_t)b * T_;

  // Q operands (A layout), head-dim chunks [0,32) and [32,64)
  const f16* qrow = qkv + (baseRow + q0 + row) * C3_ + h * D_;
  v16h aQ[2];
#pragma unroll
  for (int t = 0; t < 2; ++t) {
    v8h lo  = *(const v8h*)(qrow + 32 * t + hi * 8);
    v8h hi8 = *(const v8h*)(qrow + 32 * t + 16 + hi * 8);
    aQ[t] = hcat(lo, hi8);
  }

  float m[8], lsum[8];
  v8f acc[4];
#pragma unroll
  for (int r = 0; r < 8; ++r) { m[r] = -1.0e30f; lsum[r] = 0.0f; }
#pragma unroll
  for (int n = 0; n < 4; ++n) { v8f z = {}; acc[n] = z; }

  f16* P = ldsP[wave];
  f16* V = ldsV[wave];
  const float scale = 0.125f;  // 1/sqrt(D)
  const int qmax = q0 + 15;

  for (int kb = 0; kb <= qmax; kb += 32) {
    // ---- stage V tile rows into LDS (lane -> one key row, 128 B) ----
    {
      int vkey  = kb + lane;
      int vkeyc = vkey < T_ ? vkey : T_ - 1;
      const f16* vrow = qkv + (baseRow + vkeyc) * C3_ + 2 * C_ + h * D_;
#if ASYNC_LDS
      GV4* g = to_global_v4(vrow);
      LV4* l = to_lds_v4(V + lane * 64);
#pragma unroll
      for (int j = 0; j < 8; ++j)
        __builtin_amdgcn_global_load_async_to_lds_b128(g + j, l + j, 0, 0);
#else
#pragma unroll
      for (int j = 0; j < 8; ++j)
        *(v8h*)(V + lane * 64 + j * 8) = *(const v8h*)(vrow + j * 8);
#endif
    }

    // ---- scores: two 16-key tiles, each accumulated over head-dim 64 ----
    v8f s[2];
#pragma unroll
    for (int t = 0; t < 2; ++t) {
      int key  = kb + 16 * t + row;
      int keyc = key < T_ ? key : T_ - 1;       // clamp (masked below anyway)
      const f16* krow = qkv + (baseRow + keyc) * C3_ + C_ + h * D_;
      v8f z = {};
      s[t] = z;
#pragma unroll
      for (int c = 0; c < 2; ++c) {
        v8h lo  = *(const v8h*)(krow + 32 * c + hi * 8);
        v8h hi8 = *(const v8h*)(krow + 32 * c + 16 + hi * 8);
        s[t] = wmma16(aQ[c], hcat(lo, hi8), s[t]);
      }
    }

    // ---- online softmax (rows striped across the 8 acc VGPRs) ----
#pragma unroll
    for (int r = 0; r < 8; ++r) {
      const int q = q0 + r + 8 * hi;
      float sv0 = s[0][r] * scale;
      float sv1 = s[1][r] * scale;
      if (kb + row > q)      sv0 = -1.0e30f;    // causal mask
      if (kb + 16 + row > q) sv1 = -1.0e30f;
      float loc = fmaxf(sv0, sv1);
#pragma unroll
      for (int msk = 1; msk < 16; msk <<= 1)    // stays within 16-lane half
        loc = fmaxf(loc, __shfl_xor(loc, msk, 32));
      const float mn = fmaxf(m[r], loc);
      const float fr = __expf(m[r] - mn);
      m[r] = mn;
      const float p0 = __expf(sv0 - mn);
      const float p1 = __expf(sv1 - mn);
      P[(r + 8 * hi) * 32 + row]      = (f16)p0; // stage P for layout change
      P[(r + 8 * hi) * 32 + 16 + row] = (f16)p1;
      float ps = p0 + p1;
#pragma unroll
      for (int msk = 1; msk < 16; msk <<= 1)
        ps += __shfl_xor(ps, msk, 32);
      lsum[r] = lsum[r] * fr + ps;
#pragma unroll
      for (int n = 0; n < 4; ++n) acc[n][r] *= fr;
    }

    // ---- P back from LDS in A layout (16 rows x 32 keys) ----
    v8h plo = *(const v8h*)(P + row * 32 + hi * 8);
    v8h phi = *(const v8h*)(P + row * 32 + 16 + hi * 8);
    v16h aP = hcat(plo, phi);

#if ASYNC_LDS
    wait_async0();                 // V tile resident in LDS
#endif

    // ---- O(16x64) += P(16x32) * V(32x64), four 16-col tiles ----
#pragma unroll
    for (int n = 0; n < 4; ++n) {
      v16h bV;
#pragma unroll
      for (int e = 0; e < 16; ++e)
        bV[e] = V[((e < 8 ? 0 : 8) + hi * 8 + e) * 64 + n * 16 + row];
      acc[n] = wmma16(aP, bV, acc[n]);
    }
  }

  // ---- normalize and write out [B*T, C] ----
#pragma unroll
  for (int r = 0; r < 8; ++r) {
    const float inv = 1.0f / lsum[r];
    const size_t orow = (baseRow + q0 + r + 8 * hi) * C_ + h * D_;
#pragma unroll
    for (int n = 0; n < 4; ++n)
      att[orow + n * 16 + row] = (f16)(acc[n][r] * inv);
  }
}

// ---------------------------------------------------------------------------
extern "C" void kernel_launch(void* const* d_in, const int* in_sizes, int n_in,
                              void* d_out, int out_size, void* d_ws, size_t ws_size,
                              hipStream_t stream) {
  (void)in_sizes; (void)n_in; (void)out_size; (void)ws_size;
  const float* x      = (const float*)d_in[0];
  const float* W_attn = (const float*)d_in[1];
  const float* b_attn = (const float*)d_in[2];
  const float* W_proj = (const float*)d_in[3];
  const float* b_proj = (const float*)d_in[4];
  float* out = (float*)d_out;

  char* ws = (char*)d_ws;
  const size_t nx   = (size_t)B_ * T_ * C_;   // 4194304
  const size_t nwa  = (size_t)C_ * C3_;       // 3145728
  const size_t nwp  = (size_t)C_ * C_;        // 1048576
  const size_t nqkv = (size_t)B_ * T_ * C3_;  // 12582912

  f16* x16   = (f16*)ws; ws += nx * 2;
  f16* wa16  = (f16*)ws; ws += nwa * 2;
  f16* wp16  = (f16*)ws; ws += nwp * 2;
  f16* qkv16 = (f16*)ws; ws += nqkv * 2;
  f16* att16 = (f16*)ws;

  cvt_f16_kernel<<<(unsigned)((nx + 255) / 256), 256, 0, stream>>>(x, x16, (int)nx);
  cvt_transpose_kernel<<<(unsigned)((nwa + 255) / 256), 256, 0, stream>>>(
      W_attn, wa16, C_, C3_);
  cvt_transpose_kernel<<<(unsigned)((nwp + 255) / 256), 256, 0, stream>>>(
      W_proj, wp16, C_, C_);

  {  // qkv = x @ W_attn + b_attn  (M=B*T, N=3C, K=C)
    int M = B_ * T_, N = C3_;
    int waves = (M / 32) * (N / 64);           // 6144
    gemm_f16_kernel<<<waves / 8, 256, 0, stream>>>(x16, wa16, b_attn, qkv16,
                                                   nullptr, M, N, C_);
  }

  flash_attn_kernel<<<(B_ * H_ * (T_ / 16)) / 4, 128, 0, stream>>>(qkv16, att16);

  {  // out = att @ W_proj + b_proj  (M=B*T, N=C, K=C)
    int M = B_ * T_, N = C_;
    int waves = (M / 32) * (N / 64);           // 2048
    gemm_f16_kernel<<<waves / 8, 256, 0, stream>>>(att16, wp16, b_proj, nullptr,
                                                   out, M, N, C_);
  }
}
